// GSOrthogonal_36180804501935
// MI455X (gfx1250) — matrix operand order repaired and checked by reading, not compiled, not run on hardware
//
#include <hip/hip_runtime.h>

// ---------------------------------------------------------------------------
// GSOrthogonal (Cayley + Monarch butterfly) for MI455X / gfx1250.
//  - cayley_kernel: 128 WGs, Gauss-Jordan solve (I+skew)Y=(I-skew), Y=Q^T,
//    emits bf16 hi/lo planes pre-swizzled into WMMA B-fragment order.
//  - butterfly_kernel: 16 batch rows per WG, fully fused two-stage block GEMM
//    using split-bf16 v_wmma_f32_16x16x32_bf16 (3 products: hh, hl, lh) with
//    f32 accumulation; LDS transpose between stages; transposed coalesced out.
//  - A-fragment split uses truncation + v_perm_b32 packing (3 VALU/element)
//    so the VALU stream co-executing with the XDL WMMAs stays cheap.
// ---------------------------------------------------------------------------

typedef __attribute__((ext_vector_type(16))) __bf16       v16bf;
typedef __attribute__((ext_vector_type(8)))  float        v8f;
typedef __attribute__((ext_vector_type(4)))  float        f4v;
typedef __attribute__((ext_vector_type(4)))  unsigned int u4v;

#define N_DIM 4096
#define TB    16          // batch rows per workgroup
#define STW   4100        // y1t row stride in floats (4096 + 4 pad -> bank spread)
#define STGS  9           // staging stride for 8 l-values (+1 pad)
#define AS    132         // augmented-matrix row stride (128 + 4 pad)

__device__ __forceinline__ unsigned short f2bf(float x) {
  unsigned u = __float_as_uint(x);
  u += 0x7FFFu + ((u >> 16) & 1u);           // round-to-nearest-even
  return (unsigned short)(u >> 16);
}
__device__ __forceinline__ float bf2f(unsigned short h) {
  return __uint_as_float((unsigned)h << 16);
}

union FragW { v16bf v; unsigned u[8]; };
union FragB { v16bf v; u4v q[2]; };

__device__ __forceinline__ v8f vz8() {
  v8f z = {0.f,0.f,0.f,0.f,0.f,0.f,0.f,0.f};
  return z;
}

__device__ __forceinline__ v8f wmma_bf(v16bf a, v16bf b, v8f c) {
  return __builtin_amdgcn_wmma_f32_16x16x32_bf16(false, a, false, b,
                                                 (short)0, c, false, false);
}

// Build 16x32 bf16 A-fragment (hi/lo split) from 16 f32 values.
// Element e of lane (m + 16h) holds K = (e<8 ? 8h+e : 16+8h+(e-8)):
// a0 -> e0..3, a1 -> e4..7, a2 -> e8..11, a3 -> e12..15.
// hi = truncate-to-bf16 (residual captured exactly by lo); pack pairs with
// v_perm_b32: dst = {x1.b3, x1.b2, x0.b3, x0.b2}.
__device__ __forceinline__ void splitA(f4v a0, f4v a1, f4v a2, f4v a3,
                                       v16bf* hi, v16bf* lo) {
  float av[16];
  FragW H, L;
#pragma unroll
  for (int i = 0; i < 4; ++i) {
    av[i] = a0[i]; av[4+i] = a1[i]; av[8+i] = a2[i]; av[12+i] = a3[i];
  }
#pragma unroll
  for (int i = 0; i < 8; ++i) {
    unsigned u0 = __float_as_uint(av[2*i]);
    unsigned u1 = __float_as_uint(av[2*i+1]);
    float l0 = av[2*i]     - __uint_as_float(u0 & 0xFFFF0000u);
    float l1 = av[2*i + 1] - __uint_as_float(u1 & 0xFFFF0000u);
    H.u[i] = __builtin_amdgcn_perm(u1, u0, 0x07060302u);
    L.u[i] = __builtin_amdgcn_perm(__float_as_uint(l1), __float_as_uint(l0),
                                   0x07060302u);
  }
  *hi = H.v; *lo = L.v;
}

// Load a pre-swizzled 32x16 bf16 B-fragment: 32 contiguous bytes per lane.
__device__ __forceinline__ v16bf loadB(const unsigned short* base) {
  FragB F;
  const u4v* p = (const u4v*)base;
  F.q[0] = p[0];
  F.q[1] = p[1];
  return F.v;
}

// ---------------------------------------------------------------------------
// Cayley: solve (I + skew) Y = (I - skew), Y = Q^T, for 128 blocks
// (0..63 = R from gsoft_R, 64..127 = L from gsoft_L).  Gauss-Jordan without
// pivoting is stable here: skew ~ 0.01-scale, matrix ~ identity.
// Then write Y as bf16 hi/lo planes in WMMA B-fragment order:
//   frag element (kc, nt, lane=n+16h, e) = Y[p = 32kc+(e<8 ? 8h+e : 16+8h+e-8)][q = 16nt+n].
// ---------------------------------------------------------------------------
__global__ __launch_bounds__(128)
void cayley_kernel(const float* __restrict__ gR, const float* __restrict__ gL,
                   unsigned short* __restrict__ whi,
                   unsigned short* __restrict__ wlo) {
  __shared__ float aug[64 * AS];
  const int bid = blockIdx.x;               // 0..127
  const int c   = threadIdx.x;              // 0..127 (augmented column owner)
  const float* G = (bid < 64) ? (gR + (size_t)bid * 4096)
                              : (gL + (size_t)(bid - 64) * 4096);
  // augmented [A = I+skew | B = I-skew]
  {
    const int cc = c & 63;
    const float sgn = (c < 64) ? 1.0f : -1.0f;
    for (int i = 0; i < 64; ++i) {
      float sk = 0.5f * (G[i * 64 + cc] - G[cc * 64 + i]);
      aug[i * AS + c] = ((i == cc) ? 1.0f : 0.0f) + sgn * sk;
    }
  }
  // Gauss-Jordan; column k is frozen during step k (never read afterwards).
  for (int k = 0; k < 64; ++k) {
    __syncthreads();
    float piv = aug[k * AS + k];
    float inv = 1.0f / piv;
    __syncthreads();
    aug[k * AS + c] *= inv;                 // scale row k
    float rowk = aug[k * AS + c];           // own element, no race
    if (c != k) {
      for (int i = 0; i < 64; ++i) {
        if (i == k) continue;
        float f = aug[i * AS + k];          // column k: read-only this step
        aug[i * AS + c] -= f * rowk;
      }
    }
  }
  __syncthreads();
  // Emit swizzled bf16 hi/lo planes (RNE here; it runs once and is cheap).
  for (int fi = c; fi < 4096; fi += 128) {
    int e    = fi & 15;
    int lane = (fi >> 4) & 31;
    int n = lane & 15, h = lane >> 4;
    int nt = (fi >> 9) & 3;
    int kc = (fi >> 11) & 1;
    int p = kc * 32 + (e < 8 ? 8 * h + e : 16 + 8 * h + (e - 8));  // K index (row of Y)
    int q = nt * 16 + n;                    // N index (col of Y)
    float v = aug[p * AS + 64 + q];
    unsigned short hs = f2bf(v);
    unsigned short ls = f2bf(v - bf2f(hs));
    whi[(size_t)bid * 4096 + fi] = hs;
    wlo[(size_t)bid * 4096 + fi] = ls;
  }
}

// ---------------------------------------------------------------------------
// Fused butterfly: out[b][s*64+l] = sum_r L[l][s][r] * sum_p R[r][l][p] * x[b][r*64+p]
// ---------------------------------------------------------------------------
__global__ __launch_bounds__(256)
void butterfly_kernel(const float* __restrict__ x,
                      const unsigned short* __restrict__ whi,
                      const unsigned short* __restrict__ wlo,
                      float* __restrict__ out) {
  extern __shared__ float lds[];
  float* y1t = lds;                  // 16 x STW floats, col = q*64+k (swizzled)
  float* stg = lds + TB * STW;       // 16*64*STGS floats

  const int tid  = threadIdx.x;
  const int w    = tid >> 5;         // wave 0..7
  const int lane = tid & 31;
  const int m    = lane & 15;        // A-row / batch row within tile
  const int h    = lane >> 4;
  const int n    = lane & 15;        // B/C column within 16-tile
  const int b0   = blockIdx.x * TB;
  const float* xrow = x + (size_t)(b0 + m) * N_DIM;

  // ---------------- stage 1: y1[b][k][q] = sum_p R[k][q][p] x[b][k*64+p] ----
  for (int kb = 0; kb < 8; ++kb) {
    const int k = (w << 3) + kb;
    if (kb + 1 < 8)
      __builtin_prefetch(xrow + (k + 1) * 64, 0, 0);   // global_prefetch_b8
    v8f acc[4] = {vz8(), vz8(), vz8(), vz8()};
#pragma unroll
    for (int kc = 0; kc < 2; ++kc) {
      const int pb = k * 64 + kc * 32 + 8 * h;
      f4v a0 = __builtin_nontemporal_load((const f4v*)(xrow + pb));
      f4v a1 = __builtin_nontemporal_load((const f4v*)(xrow + pb + 4));
      f4v a2 = __builtin_nontemporal_load((const f4v*)(xrow + pb + 16));
      f4v a3 = __builtin_nontemporal_load((const f4v*)(xrow + pb + 20));
      v16bf Ah, Al;
      splitA(a0, a1, a2, a3, &Ah, &Al);
#pragma unroll
      for (int nt = 0; nt < 4; ++nt) {
        size_t wb = (size_t)k * 4096 + (size_t)(kc * 4 + nt) * 512 + lane * 16;
        v16bf Bh = loadB(whi + wb);
        v16bf Bl = loadB(wlo + wb);
        acc[nt] = wmma_bf(Ah, Bh, acc[nt]);
        acc[nt] = wmma_bf(Al, Bh, acc[nt]);
        acc[nt] = wmma_bf(Ah, Bl, acc[nt]);
      }
    }
    // Store transposed into LDS: logical col c = q*64+k, phys low6 k^(n<<2).
#pragma unroll
    for (int nt = 0; nt < 4; ++nt) {
      const int q = nt * 16 + n;
#pragma unroll
      for (int j = 0; j < 8; ++j) {
        y1t[(j + 8 * h) * STW + q * 64 + (k ^ (n << 2))] = acc[nt][j];
      }
    }
  }
  __syncthreads();

  // ---------------- stage 2: y2[b][l][s] = sum_r L[l][s][r] y1[b][r][l] -----
  for (int g = 0; g < 8; ++g) {
    const int l = g * 8 + w;
    v8f acc[4] = {vz8(), vz8(), vz8(), vz8()};
    const float* yrow = y1t + m * STW + l * 64;
    const int xsw = (l & 15) << 2;            // read-side XOR swizzle
#pragma unroll
    for (int kc = 0; kc < 2; ++kc) {
      const int r0 = kc * 32 + 8 * h;
      f4v a0 = *(const f4v*)(yrow + ((r0)      ^ xsw));
      f4v a1 = *(const f4v*)(yrow + ((r0 + 4)  ^ xsw));
      f4v a2 = *(const f4v*)(yrow + ((r0 + 16) ^ xsw));
      f4v a3 = *(const f4v*)(yrow + ((r0 + 20) ^ xsw));
      v16bf Ah, Al;
      splitA(a0, a1, a2, a3, &Ah, &Al);
#pragma unroll
      for (int nt = 0; nt < 4; ++nt) {
        size_t wb = (size_t)(64 + l) * 4096 + (size_t)(kc * 4 + nt) * 512 + lane * 16;
        v16bf Bh = loadB(whi + wb);
        v16bf Bl = loadB(wlo + wb);
        acc[nt] = wmma_bf(Ah, Bh, acc[nt]);
        acc[nt] = wmma_bf(Al, Bh, acc[nt]);
        acc[nt] = wmma_bf(Ah, Bl, acc[nt]);
      }
    }
    // Stage into LDS: stg[(row*64+s)*STGS + w]
#pragma unroll
    for (int nt = 0; nt < 4; ++nt) {
      const int s = nt * 16 + n;
#pragma unroll
      for (int j = 0; j < 8; ++j) {
        stg[((j + 8 * h) * 64 + s) * STGS + w] = acc[nt][j];
      }
    }
    __syncthreads();
    // Cooperative transposed writeout: 32B contiguous per thread.
#pragma unroll
    for (int rep = 0; rep < 4; ++rep) {
      const int idx = tid + rep * 256;            // (mm, s) pair
      const int mm = idx >> 6, s = idx & 63;
      const float* sp = stg + idx * STGS;
      f4v v0, v1;
#pragma unroll
      for (int i = 0; i < 4; ++i) { v0[i] = sp[i]; v1[i] = sp[4 + i]; }
      float* op = out + (size_t)(b0 + mm) * N_DIM + s * 64 + g * 8;
      __builtin_nontemporal_store(v0, (f4v*)op);
      __builtin_nontemporal_store(v1, (f4v*)(op + 4));
    }
    __syncthreads();
  }
}

extern "C" void kernel_launch(void* const* d_in, const int* in_sizes, int n_in,
                              void* d_out, int out_size, void* d_ws, size_t ws_size,
                              hipStream_t stream) {
  const float* x  = (const float*)d_in[0];
  const float* gR = (const float*)d_in[1];
  const float* gL = (const float*)d_in[2];
  float* out = (float*)d_out;

  unsigned short* whi = (unsigned short*)d_ws;        // 128*4096 bf16 hi plane
  unsigned short* wlo = whi + 128 * 4096;             // 128*4096 bf16 lo plane

  const int batch = in_sizes[0] / N_DIM;              // 8192
  const int grid  = batch / TB;                       // 512

  const int shmem = (TB * STW + TB * 64 * STGS) * (int)sizeof(float); // 299264 B
  (void)hipFuncSetAttribute((const void*)butterfly_kernel,
                            hipFuncAttributeMaxDynamicSharedMemorySize, shmem);

  cayley_kernel<<<128, 128, 0, stream>>>(gR, gL, whi, wlo);
  butterfly_kernel<<<grid, 256, shmem, stream>>>(x, whi, wlo, out);
}
